// GATModel_26860725469216
// MI455X (gfx1250) — matrix operand order repaired and checked by reading, not compiled, not run on hardware
//
#include <hip/hip_runtime.h>
#include <hip/hip_bf16.h>

// ---------------------------------------------------------------------------
// GAT (2-layer, heads=2 then heads=1) for MI455X / gfx1250.
// GEMMs via v_wmma_f32_16x16x32_f16 with LDS-staged B panel;
// graph phase via L2-resident f32 atomics (xh/out arrays fit in 192MB L2).
// ---------------------------------------------------------------------------

#define NN    50000
#define NE    800000
#define E2T   (NE + NN)      // edges incl. self loops
#define DIN   256
#define GK    256            // K dim of both GEMMs (256 -> 256 and 256 -> 128)

typedef __attribute__((ext_vector_type(16))) _Float16 v16h;
typedef __attribute__((ext_vector_type(8)))  float    v8f;
typedef __attribute__((ext_vector_type(4)))  float    v4f;

// ---------------- float <-> monotone uint key (for atomicMax on floats) ----
__device__ __forceinline__ unsigned f2key(float f) {
    unsigned u = __float_as_uint(f);
    return (u & 0x80000000u) ? ~u : (u | 0x80000000u);
}
__device__ __forceinline__ float key2f(unsigned k) {
    unsigned u = (k & 0x80000000u) ? (k & 0x7FFFFFFFu) : ~k;
    return __uint_as_float(u);
}

// ---------------- self-loop attr = mean of incoming edge_attr --------------
__global__ void selfloop_accum(const int* __restrict__ dstA,
                               const float* __restrict__ ea,
                               float* __restrict__ sums, float* __restrict__ cnt) {
    int e = blockIdx.x * blockDim.x + threadIdx.x;
    if (e >= NE) return;
    int d = dstA[e];
    atomicAdd(&sums[d * 2 + 0], ea[e * 2 + 0]);
    atomicAdd(&sums[d * 2 + 1], ea[e * 2 + 1]);
    atomicAdd(&cnt[d], 1.0f);
}
__global__ void selfloop_final(const float* __restrict__ sums,
                               const float* __restrict__ cnt,
                               float* __restrict__ loop_attr) {
    int n = blockIdx.x * blockDim.x + threadIdx.x;
    if (n >= NN) return;
    float c = fmaxf(cnt[n], 1.0f);
    loop_attr[n * 2 + 0] = sums[n * 2 + 0] / c;
    loop_attr[n * 2 + 1] = sums[n * 2 + 1] / c;
}

// ---------------- WMMA GEMM: C[M,N] = A[M,GK] * B[GK,N], f32 in, f16 MACs --
// Block: 256 thr = 8 waves -> 64x32 tile (wave grid 4x2), K step 32.
// B panel staged once per block into LDS in fragment-major f16 layout:
//   bpanel[((kbB*2 + wn)*32 + lane)*16 + i]  ->  one 32B read per fragment.
// OOB rows are clamped on load (never stored), keeping the unrolled inner
// loop branch-free: ds_load_b128 x2 + cvt + v_wmma per K step.
__global__ __launch_bounds__(256)
void gemm_wmma_f16(const float* __restrict__ A, const float* __restrict__ B,
                   float* __restrict__ C, int M, int N) {
    __shared__ __align__(32) _Float16 bpanel[(GK / 32) * 2 * 32 * 16]; // 16 KB

    const int t       = threadIdx.x;
    const int colBase = blockIdx.y * 32;

    // ---- cooperative B panel fill: coalesced global reads, f16 convert ----
    // pass p: rows k = p*8 .. p*8+7, each wave reads one full 32-col row.
#pragma unroll 4
    for (int p = 0; p < GK / 8; ++p) {
        int c = t & 31;
        int k = (t >> 5) + p * 8;
        float v = B[(size_t)k * N + colBase + c];
        int wn   = c >> 4;
        int lane = (c & 15) | (((k >> 3) & 1) << 4);
        int i    = (k & 7) | (((k >> 4) & 1) << 3);
        int kbB  = k >> 5;
        bpanel[(((kbB << 1) + wn) * 32 + lane) * 16 + i] = (_Float16)v;
    }
    __syncthreads();

    const int lane    = t & 31;
    const int wave    = t >> 5;
    const int waveN   = wave >> 2;
    const int tileRow = blockIdx.x * 64 + (wave & 3) * 16;
    const int tileCol = colBase + waveN * 16;
    const int hi      = lane >> 4;       // 0: lanes 0-15, 1: lanes 16-31
    const int l15     = lane & 15;

    const int rowA = tileRow + l15;
    // clamp OOB rows instead of zero-filling: their results are never stored,
    // and this keeps the unrolled pipeline free of exec-mask churn.
    const int rowLd = rowA < M ? rowA : M - 1;
    const float* arow = A + (size_t)rowLd * GK + (hi ? 8 : 0);
    // pull the tail half of this row toward the WGP before the unrolled
    // pipeline needs it (global_prefetch_b8)
    __builtin_prefetch(arow + 128, 0, 3);

    v8f acc = {};
#pragma unroll
    for (int kbB = 0; kbB < GK / 32; ++kbB) {
        // A fragment (ISA 16-bit A 16x32 layout)
        const float* ap = arow + kbB * 32;
        v4f a0 = *(const v4f*)(ap);
        v4f a1 = *(const v4f*)(ap + 4);
        v4f a2 = *(const v4f*)(ap + 16);
        v4f a3 = *(const v4f*)(ap + 20);
        v16h a;
#pragma unroll
        for (int i = 0; i < 4; ++i) {
            a[i]      = (_Float16)a0[i];
            a[4 + i]  = (_Float16)a1[i];
            a[8 + i]  = (_Float16)a2[i];
            a[12 + i] = (_Float16)a3[i];
        }
        // B fragment: one aligned 32B LDS read (2x ds_load_b128)
        v16h b = *(const v16h*)&bpanel[(((kbB << 1) + waveN) * 32 + lane) * 16];

        acc = __builtin_amdgcn_wmma_f32_16x16x32_f16(false, a, false, b,
                                                     (short)0, acc, false, false);
    }

    if (tileRow + 16 <= M) {             // full tile: unguarded stores
#pragma unroll
        for (int r = 0; r < 8; ++r)
            C[(size_t)(tileRow + r + (hi ? 8 : 0)) * N + tileCol + l15] = acc[r];
    } else {
#pragma unroll
        for (int r = 0; r < 8; ++r) {
            int row = tileRow + r + (hi ? 8 : 0);
            if (row < M) C[(size_t)row * N + tileCol + l15] = acc[r];
        }
    }
}

// ---------------- per-node attention scalars a_src/a_dst -------------------
// One wave per node; supports H in {1,2}, C = 128, D = H*C.
__global__ void node_attn(const float* __restrict__ xh,
                          const float* __restrict__ attS,
                          const float* __restrict__ attD,
                          float* __restrict__ a_src, float* __restrict__ a_dst,
                          int H, int C) {
    int gid = blockIdx.x * blockDim.x + threadIdx.x;
    int n = gid >> 5, lane = gid & 31;
    if (n >= NN) return;
    int D = H * C;
    float s0 = 0.f, s1 = 0.f, d0 = 0.f, d1 = 0.f;
    for (int col = lane; col < D; col += 32) {
        float v = xh[(size_t)n * D + col];
        float as = attS[col], ad = attD[col];
        if (col < C) { s0 += v * as; d0 += v * ad; }
        else         { s1 += v * as; d1 += v * ad; }
    }
    for (int off = 16; off; off >>= 1) {
        s0 += __shfl_xor(s0, off, 32);
        s1 += __shfl_xor(s1, off, 32);
        d0 += __shfl_xor(d0, off, 32);
        d1 += __shfl_xor(d1, off, 32);
    }
    if (lane == 0) {
        a_src[n * H + 0] = s0; a_dst[n * H + 0] = d0;
        if (H > 1) { a_src[n * H + 1] = s1; a_dst[n * H + 1] = d1; }
    }
}

// ---------------- weA[d,h] = sum_c We[d, h*C+c] * attE[h,c]  (2xH matrix) --
__global__ void edge_att_proj(const float* __restrict__ We,
                              const float* __restrict__ attE,
                              float* __restrict__ weA, int H, int C) {
    int h = threadIdx.x;
    if (h >= H) return;
    int D = H * C;
    for (int d = 0; d < 2; ++d) {
        float s = 0.f;
        for (int c = 0; c < C; ++c) s += We[d * D + h * C + c] * attE[h * C + c];
        weA[d * H + h] = s;
    }
}

// ---------------- edge logits + segment max (uint-key atomicMax) -----------
__global__ void edge_logits_k(const int* __restrict__ srcA, const int* __restrict__ dstA,
                              const float* __restrict__ ea, const float* __restrict__ loop_attr,
                              const float* __restrict__ a_src, const float* __restrict__ a_dst,
                              const float* __restrict__ weA,
                              float* __restrict__ logits, unsigned* __restrict__ mkey, int H) {
    int idx = blockIdx.x * blockDim.x + threadIdx.x;
    if (idx >= E2T * H) return;
    int e = idx / H, h = idx - e * H;
    int s, d; float e0, e1;
    if (e < NE) { s = srcA[e]; d = dstA[e]; e0 = ea[e * 2]; e1 = ea[e * 2 + 1]; }
    else        { s = d = e - NE; e0 = loop_attr[s * 2]; e1 = loop_attr[s * 2 + 1]; }
    float a = a_src[s * H + h] + a_dst[d * H + h] + e0 * weA[h] + e1 * weA[H + h];
    float lr = a > 0.f ? a : 0.2f * a;                 // leaky_relu(0.2)
    logits[idx] = lr;
    atomicMax(&mkey[d * H + h], f2key(lr));
}

// ---------------- exp(logit - max) + segment sum ---------------------------
__global__ void edge_exp_k(const int* __restrict__ dstA,
                           float* __restrict__ logits, const unsigned* __restrict__ mkey,
                           float* __restrict__ denom, int H) {
    int idx = blockIdx.x * blockDim.x + threadIdx.x;
    if (idx >= E2T * H) return;
    int e = idx / H, h = idx - e * H;
    int d = (e < NE) ? dstA[e] : e - NE;
    float ex = expf(logits[idx] - key2f(mkey[d * H + h]));
    logits[idx] = ex;                                  // reuse as alpha numerator
    atomicAdd(&denom[d * H + h], ex);
}

// ---------------- message scatter: out[dst] += alpha * xh[src] -------------
// One wave per edge; each lane owns D/32 contiguous columns (within one head).
__global__ void scatter_msgs(const float* __restrict__ xh,
                             const float* __restrict__ alpha, const float* __restrict__ denom,
                             const int* __restrict__ srcA, const int* __restrict__ dstA,
                             float* __restrict__ out, int H, int C) {
    int gid = blockIdx.x * blockDim.x + threadIdx.x;
    int e = gid >> 5, lane = gid & 31;
    if (e >= E2T) return;
    int s, d;
    if (e < NE) { s = srcA[e]; d = dstA[e]; } else { s = d = e - NE; }
    int D = H * C;
    int vpl = D >> 5;                     // 8 (D=256) or 4 (D=128)
    int c0 = lane * vpl;
    int h = c0 / C;
    float w = alpha[e * H + h] / (denom[d * H + h] + 1e-16f);
    const float* xp = xh + (size_t)s * D + c0;
    float* op = out + (size_t)d * D + c0;
    for (int i = 0; i < vpl; ++i) atomicAdd(op + i, w * xp[i]);
}

// ---------------- bias (+ optional ELU), elementwise, in/out may alias -----
__global__ void bias_act_k(const float* __restrict__ acc, const float* __restrict__ bias,
                           float* __restrict__ out, int total, int D, int elu) {
    int i = blockIdx.x * blockDim.x + threadIdx.x;
    if (i >= total) return;
    float v = acc[i] + bias[i % D];
    if (elu) v = v > 0.f ? v : (expf(v) - 1.f);
    out[i] = v;
}

// ---------------------------------------------------------------------------
extern "C" void kernel_launch(void* const* d_in, const int* in_sizes, int n_in,
                              void* d_out, int out_size, void* d_ws, size_t ws_size,
                              hipStream_t stream) {
    const float* x     = (const float*)d_in[0];
    const int*   eidx  = (const int*)d_in[1];       // [2, NE] flat
    const float* eattr = (const float*)d_in[2];
    const float* W1    = (const float*)d_in[3];
    const float* attS1 = (const float*)d_in[4];
    const float* attD1 = (const float*)d_in[5];
    const float* We1   = (const float*)d_in[6];
    const float* attE1 = (const float*)d_in[7];
    const float* b1    = (const float*)d_in[8];
    const float* W2    = (const float*)d_in[9];
    const float* attS2 = (const float*)d_in[10];
    const float* attD2 = (const float*)d_in[11];
    const float* We2   = (const float*)d_in[12];
    const float* attE2 = (const float*)d_in[13];
    const float* b2    = (const float*)d_in[14];
    float* out = (float*)d_out;

    const int* srcA = eidx;
    const int* dstA = eidx + NE;

    // ---- carve workspace ----
    char* w = (char*)d_ws;
    size_t off = 0;
    auto carve = [&](size_t bytes) -> void* {
        void* p = w + off;
        off = (off + bytes + 255) & ~(size_t)255;
        return p;
    };
    float*    loop_sum  = (float*)carve((size_t)NN * 2 * 4);
    float*    loop_cnt  = (float*)carve((size_t)NN * 4);
    float*    loop_attr = (float*)carve((size_t)NN * 2 * 4);
    float*    weA1      = (float*)carve(64);
    float*    weA2      = (float*)carve(64);
    float*    xh1       = (float*)carve((size_t)NN * 256 * 4);
    float*    asrc1     = (float*)carve((size_t)NN * 2 * 4);
    float*    adst1     = (float*)carve((size_t)NN * 2 * 4);
    float*    logit1    = (float*)carve((size_t)E2T * 2 * 4);
    unsigned* mkey1     = (unsigned*)carve((size_t)NN * 2 * 4);
    float*    den1      = (float*)carve((size_t)NN * 2 * 4);
    float*    hbuf      = (float*)carve((size_t)NN * 256 * 4);   // conv1 acc -> ELU -> h
    float*    xh2       = (float*)carve((size_t)NN * 128 * 4);
    float*    asrc2     = (float*)carve((size_t)NN * 4);
    float*    adst2     = (float*)carve((size_t)NN * 4);
    float*    logit2    = (float*)carve((size_t)E2T * 4);
    unsigned* mkey2     = (unsigned*)carve((size_t)NN * 4);
    float*    den2      = (float*)carve((size_t)NN * 4);
    (void)ws_size; (void)in_sizes; (void)n_in;

    const int TB = 256;
    auto blocks = [](long long t, int b) { return (unsigned)((t + b - 1) / b); };

    // ---- phase 0: self-loop edge_attr (mean over incoming edges) ----
    hipMemsetAsync(loop_sum, 0, (size_t)NN * 2 * 4, stream);
    hipMemsetAsync(loop_cnt, 0, (size_t)NN * 4, stream);
    selfloop_accum<<<blocks(NE, TB), TB, 0, stream>>>(dstA, eattr, loop_sum, loop_cnt);
    selfloop_final<<<blocks(NN, TB), TB, 0, stream>>>(loop_sum, loop_cnt, loop_attr);

    // ================= layer 1: H=2, C=128, Din=256 =================
    {
        const int H = 2, C = 128, D = H * C;
        dim3 g(blocks(NN, 64), D / 32);
        gemm_wmma_f16<<<g, TB, 0, stream>>>(x, W1, xh1, NN, D);
        node_attn<<<blocks((long long)NN * 32, TB), TB, 0, stream>>>(xh1, attS1, attD1,
                                                                     asrc1, adst1, H, C);
        edge_att_proj<<<1, 32, 0, stream>>>(We1, attE1, weA1, H, C);
        hipMemsetAsync(mkey1, 0, (size_t)NN * H * 4, stream);
        hipMemsetAsync(den1, 0, (size_t)NN * H * 4, stream);
        hipMemsetAsync(hbuf, 0, (size_t)NN * D * 4, stream);
        edge_logits_k<<<blocks((long long)E2T * H, TB), TB, 0, stream>>>(
            srcA, dstA, eattr, loop_attr, asrc1, adst1, weA1, logit1, mkey1, H);
        edge_exp_k<<<blocks((long long)E2T * H, TB), TB, 0, stream>>>(
            dstA, logit1, mkey1, den1, H);
        scatter_msgs<<<blocks((long long)E2T * 32, TB), TB, 0, stream>>>(
            xh1, logit1, den1, srcA, dstA, hbuf, H, C);
        bias_act_k<<<blocks((long long)NN * D, TB), TB, 0, stream>>>(
            hbuf, b1, hbuf, NN * D, D, /*elu=*/1);
    }

    // ================= layer 2: H=1, C=128, Din=256 =================
    {
        const int H = 1, C = 128, D = H * C;
        dim3 g(blocks(NN, 64), D / 32);
        gemm_wmma_f16<<<g, TB, 0, stream>>>(hbuf, W2, xh2, NN, D);
        node_attn<<<blocks((long long)NN * 32, TB), TB, 0, stream>>>(xh2, attS2, attD2,
                                                                     asrc2, adst2, H, C);
        edge_att_proj<<<1, 32, 0, stream>>>(We2, attE2, weA2, H, C);
        hipMemsetAsync(mkey2, 0, (size_t)NN * H * 4, stream);
        hipMemsetAsync(den2, 0, (size_t)NN * H * 4, stream);
        hipMemsetAsync(out, 0, (size_t)out_size * 4, stream);
        edge_logits_k<<<blocks((long long)E2T * H, TB), TB, 0, stream>>>(
            srcA, dstA, eattr, loop_attr, asrc2, adst2, weA2, logit2, mkey2, H);
        edge_exp_k<<<blocks((long long)E2T * H, TB), TB, 0, stream>>>(
            dstA, logit2, mkey2, den2, H);
        scatter_msgs<<<blocks((long long)E2T * 32, TB), TB, 0, stream>>>(
            xh2, logit2, den2, srcA, dstA, out, H, C);
        bias_act_k<<<blocks((long long)NN * D, TB), TB, 0, stream>>>(
            out, b2, out, NN * D, D, /*elu=*/0);
    }
}
